// LinearSelfAttention_9646496546949
// MI455X (gfx1250) — matrix operand order repaired and checked by reading, not compiled, not run on hardware
//
#include <hip/hip_runtime.h>

typedef __attribute__((ext_vector_type(16))) __bf16 v16bf;
typedef __attribute__((ext_vector_type(8)))  __bf16 v8bf;
typedef __attribute__((ext_vector_type(8)))  float  v8f;

#define BATCH 8
#define SEQ   4096
#define DIM   512
#define ROWS  (BATCH * SEQ)           // 32768
#define QSCALE 0.04419417382415922f   // 1/sqrt(512)
#define KCH   256                     // K-chunk for the kv stage (TDM tile)
#define NCH   (SEQ / KCH)             // 16 chunks

#if __has_builtin(__builtin_amdgcn_tensor_load_to_lds) && \
    __has_builtin(__builtin_amdgcn_s_wait_tensorcnt)
#define USE_TDM 1
#else
#define USE_TDM 0
#endif

// ---------------- WMMA fragment helpers (ISA 7.12.2 layouts) ----------------
// A fragment: 16x32 bf16 (MxK). lane: m = lane&15 row, g = lane>>4.
// halves e<8 -> K = 8g+e ; e>=8 -> K = 8g+e+8  => two contiguous 16B runs
// at half-offset (kk + 8g) and (kk + 8g + 16).  Served from LDS (ds_load_b128).
__device__ __forceinline__ v16bf frag_a_lds(const __bf16* __restrict__ As,
                                            int ldh, int m, int g, int kk) {
  const __bf16* p = As + m * ldh + kk + 8 * g;
  const v8bf lo = *(const v8bf*)p;
  const v8bf hi = *(const v8bf*)(p + 16);
  return __builtin_shufflevector(lo, hi, 0, 1, 2, 3, 4, 5, 6, 7,
                                 8, 9, 10, 11, 12, 13, 14, 15);
}
// B fragment: 32x16 bf16 (KxN) from a B^T ([col][K]) layout: lane col n=lane&15,
// halves e<8 -> K = 8g+e ; e>=8 -> K = 8g+e+8  => two 16B runs at (kk + 8g), (kk + 8g + 16).
__device__ __forceinline__ v16bf frag_b_g(const __bf16* __restrict__ row,
                                          int g, int kk) {
  const __bf16* p = row + kk + 8 * g;
  const v8bf lo = *(const v8bf*)p;
  const v8bf hi = *(const v8bf*)(p + 16);
  return __builtin_shufflevector(lo, hi, 0, 1, 2, 3, 4, 5, 6, 7,
                                 8, 9, 10, 11, 12, 13, 14, 15);
}
__device__ __forceinline__ v8f wmma_bf16(v16bf a, v16bf b, v8f c) {
  return __builtin_amdgcn_wmma_f32_16x16x32_bf16(false, a, false, b, (short)0, c,
                                                 false, false);
}

// ------------------------- TDM (Tensor Data Mover) --------------------------
#if USE_TDM
typedef __attribute__((ext_vector_type(4))) unsigned int v4u;
typedef __attribute__((ext_vector_type(8))) int v8i_;
typedef __attribute__((ext_vector_type(4))) int v4i_;

__device__ __forceinline__ unsigned int lds_offset_of(const __bf16* p) {
  return (unsigned int)(size_t)(__attribute__((address_space(3))) const __bf16*)p;
}
// 2D bf16 tile load: tile_d0 x tile_d1 elements, row stride = stride_elems.
// D# per CDNA5 ISA (group0: count|lds|global|type=2,
// group1: data_size=2B, tensor dims, tile dims, dim0 stride).
__device__ __forceinline__ void tdm_load_2d_bf16(unsigned int lds_byte,
                                                 const void* gaddr,
                                                 int tile_d0, int tile_d1,
                                                 int tensor_d0, int tensor_d1,
                                                 int stride_elems) {
  const unsigned long long ga = (unsigned long long)(size_t)gaddr;
  v4u g0;
  g0[0] = 1u;                                           // count=1 (valid, user)
  g0[1] = lds_byte;                                     // lds_addr
  g0[2] = (unsigned int)ga;                             // global_addr[31:0]
  g0[3] = (unsigned int)((ga >> 32) & 0x1FFFFFFu) | (2u << 30);  // [56:32]|type=2
  v8i_ g1;
  g1[0] = (int)(1u << 16);                              // data_size=1 (2 bytes)
  g1[1] = (int)(((unsigned)tensor_d0 & 0xFFFFu) << 16); // tensor_dim0[15:0]
  g1[2] = (int)(((unsigned)tensor_d0 >> 16) | (((unsigned)tensor_d1 & 0xFFFFu) << 16));
  g1[3] = (int)(((unsigned)tensor_d1 >> 16) | ((unsigned)tile_d0 << 16));
  g1[4] = (int)((unsigned)tile_d1);                     // tile_dim1 | tile_dim2=0
  g1[5] = (int)((unsigned)stride_elems);                // tensor_dim0_stride[31:0]
  g1[6] = 0;
  g1[7] = 0;
  const v4i_ z4 = {0, 0, 0, 0};
  const v8i_ z8 = {0, 0, 0, 0, 0, 0, 0, 0};
  // 6-arg form (clang-23 / therock-10.0 headers): (g0, g1, g2, g3, g4, cpol)
  __builtin_amdgcn_tensor_load_to_lds(g0, g1, z4, z4, z8, 0);
}
#endif

// ---------------------------------------------------------------------------
// Kernel 0: convert Wq|Wk|Wv (fp32, each [512][512] as [out][in]) into one
// fused bf16 buffer W16[1536][512] so stage-1 B fragments are pure b128 loads.
// ---------------------------------------------------------------------------
__global__ __launch_bounds__(256) void wcvt_kernel(
    const float* __restrict__ Wq, const float* __restrict__ Wk,
    const float* __restrict__ Wv, __bf16* __restrict__ W16) {
  const size_t i = ((size_t)blockIdx.x * 256 + threadIdx.x) * 4;   // 786432 total
  const size_t row = i >> 9;
  const float* W = (row < 512) ? Wq : ((row < 1024) ? Wk : Wv);
  const float4 f = *(const float4*)(W + ((row & 511) << 9) + (i & 511));
  W16[i + 0] = (__bf16)f.x;
  W16[i + 1] = (__bf16)f.y;
  W16[i + 2] = (__bf16)f.z;
  W16[i + 3] = (__bf16)f.w;
}

// ---------------------------------------------------------------------------
// Kernel 1: fused QKV projection + bias + rotary + scale/weights.
// One block per 16-row strip: phi strip converted to bf16 in LDS once, then
// 8 waves sweep 64 column-tile jobs (Q/K jobs own the paired (d, d+256) tiles
// so rotary is applied in-register).  q row-major bf16; k,v transposed
// [b][channel][n] bf16 for stage 2.
// ---------------------------------------------------------------------------
__global__ __launch_bounds__(256) void qkv_rotary_kernel(
    const float* __restrict__ phi, const float* __restrict__ coords,
    const float* __restrict__ weights, const __bf16* __restrict__ W16,
    const float* __restrict__ bq, const float* __restrict__ bk,
    const float* __restrict__ bv, const float* __restrict__ Wrot,
    __bf16* __restrict__ q16,    // [ROWS][DIM]
    __bf16* __restrict__ k16t,   // [BATCH][DIM][SEQ]
    __bf16* __restrict__ v16t)   // [BATCH][DIM][SEQ]
{
  __shared__ __bf16 Atile[16 * DIM];   // 16 KB

  const int tid  = threadIdx.x;
  const int lane = tid & 31;
  const int wave = tid >> 5;
  const int m = lane & 15, g = lane >> 4;
  const int row_base = blockIdx.x * 16;

  // ---- stage phi strip (fp32 -> bf16) into LDS: 32 floats per thread ----
  {
    const int base = tid * 32;
    const int row = base >> 9, col = base & 511;
    const float* src = phi + (size_t)(row_base + row) * DIM + col;
#pragma unroll
    for (int i = 0; i < 32; i += 4) {
      const float4 f = *(const float4*)(src + i);
      Atile[row * DIM + col + i + 0] = (__bf16)f.x;
      Atile[row * DIM + col + i + 1] = (__bf16)f.y;
      Atile[row * DIM + col + i + 2] = (__bf16)f.z;
      Atile[row * DIM + col + i + 3] = (__bf16)f.w;
    }
  }
  __syncthreads();

  // ---- cache per-row coords / weights for this lane's 8 output rows ----
  float cx[8], cy[8], cz[8], wt[8];
#pragma unroll
  for (int r = 0; r < 8; ++r) {
    const int row = row_base + (g << 3) + r;
    cx[r] = coords[row * 3 + 0];
    cy[r] = coords[row * 3 + 1];
    cz[r] = coords[row * 3 + 2];
    wt[r] = weights[row];
  }

  // ---- 8 jobs per wave over the fused 1536-wide output ----
  for (int jt = 0; jt < 8; ++jt) {
    const int job = jt * 8 + wave;        // wave-uniform, 0..63
    if (job < 32) {                       // Q (job<16) / K paired tiles
      const bool isQ = job < 16;
      const int dtile = (job & 15) * 16;  // 0..240
      const int wrow = (isQ ? 0 : 512) + dtile;
      const __bf16* B0row = W16 + (size_t)(wrow + m) * DIM;
      const __bf16* B1row = W16 + (size_t)(wrow + 256 + m) * DIM;
      v8f c0 = {}, c1 = {};
#pragma unroll 4
      for (int kk = 0; kk < DIM; kk += 32) {
        const v16bf a = frag_a_lds(Atile, DIM, m, g, kk);
        c0 = wmma_bf16(a, frag_b_g(B0row, g, kk), c0);
        c1 = wmma_bf16(a, frag_b_g(B1row, g, kk), c1);
      }
      const int col = dtile + m;          // 0..255
      const float w0 = Wrot[col * 3 + 0];
      const float w1 = Wrot[col * 3 + 1];
      const float w2 = Wrot[col * 3 + 2];
      const float* bias = isQ ? bq : bk;
      const float bl = bias[col], bh = bias[col + 256];
#pragma unroll
      for (int r = 0; r < 8; ++r) {
        const int row = row_base + (g << 3) + r;
        const int b  = row >> 12;
        const int nn = row & (SEQ - 1);
        float sn, cs;
        __sincosf(cx[r] * w0 + cy[r] * w1 + cz[r] * w2, &sn, &cs);
        float qa = c0[r] + bl;
        float qb = c1[r] + bh;
        if (isQ) { qa *= QSCALE; qb *= QSCALE; }
        const float lo = qa * cs - qb * sn;
        const float hi = qa * sn + qb * cs;
        if (isQ) {
          q16[(size_t)row * DIM + col]       = (__bf16)lo;
          q16[(size_t)row * DIM + col + 256] = (__bf16)hi;
        } else {
          k16t[((size_t)b * DIM + col) * SEQ + nn]       = (__bf16)lo;
          k16t[((size_t)b * DIM + col + 256) * SEQ + nn] = (__bf16)hi;
        }
      }
    } else {                              // V tile + quadrature weights
      const int ctile = (job - 32) * 16;  // 0..496
      const __bf16* Brow = W16 + (size_t)(1024 + ctile + m) * DIM;
      v8f c0 = {};
#pragma unroll 4
      for (int kk = 0; kk < DIM; kk += 32) {
        const v16bf a = frag_a_lds(Atile, DIM, m, g, kk);
        c0 = wmma_bf16(a, frag_b_g(Brow, g, kk), c0);
      }
      const int col = ctile + m;
      const float bl = bv[col];
#pragma unroll
      for (int r = 0; r < 8; ++r) {
        const int row = row_base + (g << 3) + r;
        const int b  = row >> 12;
        const int nn = row & (SEQ - 1);
        v16t[((size_t)b * DIM + col) * SEQ + nn] = (__bf16)(wt[r] * (c0[r] + bl));
      }
    }
  }
}

// ---------------------------------------------------------------------------
// Kernel 2: kv[b,d,e] = sum_n k[b,n,d] * v[b,n,e].  One block per (b, d-tile);
// the 16xKCH A-chunk of k^T is staged into LDS by the Tensor Data Mover
// (double-buffered, wave0 issues, s_wait_tensorcnt + barrier publishes it),
// and each wave owns 4 e-tiles so every A fragment feeds 4 WMMAs.
// Stores kv transposed kvT[b][e][d] bf16 (contiguous in d).
// ---------------------------------------------------------------------------
__global__ __launch_bounds__(256) void kv_kernel(
    const __bf16* __restrict__ k16t, const __bf16* __restrict__ v16t,
    __bf16* __restrict__ kvT)   // [BATCH][DIM][DIM] = kvT[b][e][d]
{
  __shared__ __bf16 Abuf[2][16 * KCH];   // 2 x 8 KB

  const int tid  = threadIdx.x;
  const int lane = tid & 31;
  const int wave = tid >> 5;
  const int m = lane & 15, g = lane >> 4;
  const int b     = blockIdx.x >> 5;          // 0..7
  const int dtile = (blockIdx.x & 31) * 16;   // 0..496

  const __bf16* Abase = k16t + ((size_t)b * DIM + dtile) * SEQ;  // block-uniform
  const __bf16* Brow[4];
#pragma unroll
  for (int tj = 0; tj < 4; ++tj)
    Brow[tj] = v16t + ((size_t)b * DIM + (wave + tj * 8) * 16 + m) * SEQ;

  v8f acc[4] = {v8f{}, v8f{}, v8f{}, v8f{}};

  // ---- stage helpers ----
  auto stage_issue = [&](int buf, int ch) {
#if USE_TDM
    if (wave == 0) {
      tdm_load_2d_bf16(lds_offset_of(Abuf[buf]), Abase + (size_t)ch * KCH,
                       /*tile*/ KCH, 16, /*tensor*/ SEQ, DIM, /*stride*/ SEQ);
    }
#else
    const int base = tid * 16;
    const int row = base >> 8;          // / KCH
    const int col = base & (KCH - 1);
    const __bf16* src = Abase + (size_t)row * SEQ + (size_t)ch * KCH + col;
    __bf16* dst = Abuf[buf] + row * KCH + col;
    *(v8bf*)dst = *(const v8bf*)src;
    *(v8bf*)(dst + 8) = *(const v8bf*)(src + 8);
#endif
  };
  auto stage_fence = [&]() {
#if USE_TDM
    if (wave == 0) __builtin_amdgcn_s_wait_tensorcnt(0);
#endif
  };

  stage_issue(0, 0);
  stage_fence();
  __syncthreads();

  for (int ch = 0; ch < NCH; ++ch) {
    if (ch + 1 < NCH) stage_issue((ch + 1) & 1, ch + 1);   // overlap DMA
    const __bf16* As = Abuf[ch & 1];
    const int koff = ch * KCH;
#pragma unroll 2
    for (int kk = 0; kk < KCH; kk += 32) {
      const v16bf a = frag_a_lds(As, KCH, m, g, kk);
#pragma unroll
      for (int tj = 0; tj < 4; ++tj)
        acc[tj] = wmma_bf16(a, frag_b_g(Brow[tj], g, koff + kk), acc[tj]);
    }
    stage_fence();
    __syncthreads();
  }

#pragma unroll
  for (int tj = 0; tj < 4; ++tj) {
    const int e = (wave + tj * 8) * 16 + m;
    __bf16* dst = kvT + ((size_t)b * DIM + e) * DIM + dtile + (g << 3);
#pragma unroll
    for (int r = 0; r < 8; ++r) dst[r] = (__bf16)acc[tj][r];
  }
}

// ---------------------------------------------------------------------------
// Kernel 3: out[b,n,e] = sum_d q[b,n,d] * kv[b,d,e].  One block per 16-row
// strip: q strip staged in LDS once, 8 waves x 4 e-tiles cover all 512 cols.
// kvT (512 KB/batch) stays hot in L2.  fp32 output.
// ---------------------------------------------------------------------------
__global__ __launch_bounds__(256) void out_kernel(
    const __bf16* __restrict__ q16, const __bf16* __restrict__ kvT,
    float* __restrict__ out)    // [ROWS][DIM]
{
  __shared__ __bf16 Atile[16 * DIM];   // 16 KB

  const int tid  = threadIdx.x;
  const int lane = tid & 31;
  const int wave = tid >> 5;
  const int m = lane & 15, g = lane >> 4;
  const int row_base = blockIdx.x * 16;
  const int b = row_base >> 12;

  {  // stage q strip (bf16 copy): 32 halves per thread
    const int base = tid * 32;
    const int row = base >> 9, col = base & 511;
    const __bf16* src = q16 + (size_t)(row_base + row) * DIM + col;
    __bf16* dst = Atile + row * DIM + col;
#pragma unroll
    for (int i = 0; i < 32; i += 8)
      *(v8bf*)(dst + i) = *(const v8bf*)(src + i);
  }
  __syncthreads();

  for (int et = 0; et < 4; ++et) {
    const int etile = (wave + et * 8) * 16;
    const __bf16* Brow = kvT + ((size_t)b * DIM + etile + m) * DIM;
    v8f c = {};
#pragma unroll 4
    for (int kk = 0; kk < DIM; kk += 32)
      c = wmma_bf16(frag_a_lds(Atile, DIM, m, g, kk), frag_b_g(Brow, g, kk), c);
    const int e = etile + m;
#pragma unroll
    for (int r = 0; r < 8; ++r)
      out[(size_t)(row_base + (g << 3) + r) * DIM + e] = c[r];
  }
}

// ---------------------------------------------------------------------------
extern "C" void kernel_launch(void* const* d_in, const int* in_sizes, int n_in,
                              void* d_out, int out_size, void* d_ws, size_t ws_size,
                              hipStream_t stream) {
  (void)in_sizes; (void)n_in; (void)out_size; (void)ws_size;
  const float* phi     = (const float*)d_in[0];
  const float* coords  = (const float*)d_in[1];
  const float* weights = (const float*)d_in[2];
  const float* Wq      = (const float*)d_in[3];
  const float* bq      = (const float*)d_in[4];
  const float* Wk      = (const float*)d_in[5];
  const float* bk      = (const float*)d_in[6];
  const float* Wv      = (const float*)d_in[7];
  const float* bv      = (const float*)d_in[8];
  const float* Wrot    = (const float*)d_in[9];
  float* out = (float*)d_out;

  char* ws = (char*)d_ws;
  size_t off = 0;
  __bf16* q16  = (__bf16*)(ws + off); off += (size_t)ROWS * DIM * sizeof(__bf16);
  __bf16* k16t = (__bf16*)(ws + off); off += (size_t)ROWS * DIM * sizeof(__bf16);
  __bf16* v16t = (__bf16*)(ws + off); off += (size_t)ROWS * DIM * sizeof(__bf16);
  __bf16* kvT  = (__bf16*)(ws + off); off += (size_t)BATCH * DIM * DIM * sizeof(__bf16);
  __bf16* W16  = (__bf16*)(ws + off); off += (size_t)3 * DIM * DIM * sizeof(__bf16);

  wcvt_kernel<<<dim3((3 * DIM * DIM) / (256 * 4)), 256, 0, stream>>>(Wq, Wk, Wv, W16);
  qkv_rotary_kernel<<<dim3(ROWS / 16), 256, 0, stream>>>(
      phi, coords, weights, W16, bq, bk, bv, Wrot, q16, k16t, v16t);
  kv_kernel<<<dim3(BATCH * 32), 256, 0, stream>>>(k16t, v16t, kvT);
  out_kernel<<<dim3(ROWS / 16), 256, 0, stream>>>(q16, kvT, out);
}
